// GNN_1125281431593
// MI455X (gfx1250) — matrix operand order repaired and checked by reading, compiled, not run on hardware
//
#include <hip/hip_runtime.h>
#include <math.h>

typedef __bf16 v16bf __attribute__((ext_vector_type(16)));
typedef float  v8f   __attribute__((ext_vector_type(8)));

#define FEAT 64
#define OUTC 16

__device__ __forceinline__ unsigned short f32_to_bf16_rne(float f) {
    unsigned int u = __float_as_uint(f);
    u += 0x7FFFu + ((u >> 16) & 1u);
    return (unsigned short)(u >> 16);
}

// ---------------------------------------------------------------------------
// gcn_norm helpers
// ---------------------------------------------------------------------------
__global__ void __launch_bounds__(256) zero_kernel(float* __restrict__ p, int n) {
    int t = blockIdx.x * blockDim.x + threadIdx.x;
    if (t < n) p[t] = 0.0f;
}

__global__ void __launch_bounds__(256) deg_kernel(const int* __restrict__ rows,
                                                  float* __restrict__ deg, int nedges) {
    int t = blockIdx.x * blockDim.x + threadIdx.x;
    if (t < nedges) atomicAdd(&deg[rows[t]], 1.0f);
}

__global__ void __launch_bounds__(256) dinv_kernel(float* __restrict__ d, int n) {
    int t = blockIdx.x * blockDim.x + threadIdx.x;
    if (t < n) d[t] = rsqrtf(d[t] + 1.0f);   // +1 = self loop; always > 0
}

// ---------------------------------------------------------------------------
// Weight prep: write W (fp32, row-major [K x Ncols]) as bf16 WMMA B-fragments.
// Fragment f = nt*nkt + kt; element (f, lane, h):
//   k = kt*32 + ((lane<16)?0:8) + (h<8 ? h : 16 + (h-8)) ,  col = nt*16 + (lane&15)
// Stored contiguously: Wfrag[(f*32 + lane)*16 + h] -> one aligned v16bf per lane.
// ---------------------------------------------------------------------------
__global__ void __launch_bounds__(256) prep_wfrag_kernel(const float* __restrict__ W,
                                                         unsigned short* __restrict__ Wfrag,
                                                         int ncols, int nkt, int total) {
    int t = blockIdx.x * blockDim.x + threadIdx.x;
    if (t >= total) return;
    int h    = t & 15;
    int lane = (t >> 4) & 31;
    int f    = t >> 9;
    int kt   = f % nkt;
    int nt   = f / nkt;
    int k    = kt * 32 + ((lane < 16) ? 0 : 8) + ((h < 8) ? h : 16 + (h - 8));
    int col  = nt * 16 + (lane & 15);
    Wfrag[t] = f32_to_bf16_rne(W[k * ncols + col]);
}

// ---------------------------------------------------------------------------
// SpMM hop: y = dinv_i^2 * x_i  (self loop)  then scatter edges with atomics
// ---------------------------------------------------------------------------
__global__ void __launch_bounds__(256) selfloop_init_kernel(const float* __restrict__ xin,
                                                            const float* __restrict__ dinv,
                                                            float* __restrict__ yout, int total) {
    int t = blockIdx.x * blockDim.x + threadIdx.x;
    if (t >= total) return;
    int i = t >> 6;                            // /FEAT
    float d = dinv[i];
    yout[t] = d * d * xin[t];
}

// one wave per edge; 32 lanes x float2 = 64 features, coalesced row read.
// Edge index forced scalar so rows/cols/dinv lower to SMEM loads.
__global__ void __launch_bounds__(256) spmm_scatter_kernel(const int* __restrict__ rows,
                                                           const int* __restrict__ cols,
                                                           const float* __restrict__ dinv,
                                                           const float* __restrict__ xin,
                                                           float* __restrict__ yout, int nedges) {
    const int e    = (int)blockIdx.x * 8 + ((int)threadIdx.x >> 5);
    const int lane = threadIdx.x & 31;
    if (e >= nedges) return;
    const int eu = __builtin_amdgcn_readfirstlane(e);   // wave-uniform -> scalar loads
    const int r  = rows[eu];
    const int c  = cols[eu];
    const float w = dinv[r] * dinv[c];
    const float2 xv = *(const float2*)(xin + (size_t)c * FEAT + lane * 2);
    atomicAdd(yout + (size_t)r * FEAT + lane * 2,     w * xv.x);
    atomicAdd(yout + (size_t)r * FEAT + lane * 2 + 1, w * xv.y);
}

// ---------------------------------------------------------------------------
// A fragment: 16x32 bf16, lanes 0-15 row M=lane, K={0..7,16..23}; lanes 16-31: +8
// ---------------------------------------------------------------------------
__device__ __forceinline__ v16bf load_a_frag(const float* __restrict__ rp, int kb8) {
    float4 f0 = *(const float4*)(rp + kb8);
    float4 f1 = *(const float4*)(rp + kb8 + 4);
    float4 f2 = *(const float4*)(rp + kb8 + 16);
    float4 f3 = *(const float4*)(rp + kb8 + 20);
    v16bf a;
    a[0]  = (__bf16)f0.x; a[1]  = (__bf16)f0.y; a[2]  = (__bf16)f0.z; a[3]  = (__bf16)f0.w;
    a[4]  = (__bf16)f1.x; a[5]  = (__bf16)f1.y; a[6]  = (__bf16)f1.z; a[7]  = (__bf16)f1.w;
    a[8]  = (__bf16)f2.x; a[9]  = (__bf16)f2.y; a[10] = (__bf16)f2.z; a[11] = (__bf16)f2.w;
    a[12] = (__bf16)f3.x; a[13] = (__bf16)f3.y; a[14] = (__bf16)f3.z; a[15] = (__bf16)f3.w;
    return a;
}

__device__ __forceinline__ v16bf load_b_frag(const unsigned short* __restrict__ Wfrag,
                                             int frag, int lane) {
    return *(const v16bf*)(Wfrag + ((size_t)(frag * 32 + lane) << 4));
}

// ---------------------------------------------------------------------------
// GEMM1: H = SELU(P @ W1 + b1), P:[N,64] fp32; 8 WMMAs per wave (16 rows)
// ---------------------------------------------------------------------------
__global__ void __launch_bounds__(256) gemm1_selu_kernel(const float* __restrict__ P,
                                                         const unsigned short* __restrict__ Wfrag,
                                                         const float* __restrict__ bias,
                                                         float* __restrict__ H, int nrows) {
    const int wave = (int)blockIdx.x * ((int)blockDim.x >> 5) + ((int)threadIdx.x >> 5);
    const int lane = threadIdx.x & 31;
    const int row0 = wave << 4;
    if (row0 >= nrows) return;                 // wave-uniform: EXEC stays all-ones
    const int m   = lane & 15;
    const int kb8 = (lane < 16) ? 0 : 8;
    const int mb  = (lane < 16) ? 0 : 8;

    v8f acc[4] = {};
#pragma unroll
    for (int kt = 0; kt < 2; ++kt) {
        const float* rp = P + (size_t)(row0 + m) * FEAT + kt * 32;
        v16bf a = load_a_frag(rp, kb8);
#pragma unroll
        for (int nt = 0; nt < 4; ++nt) {
            v16bf b = load_b_frag(Wfrag, nt * 2 + kt, lane);
            acc[nt] = __builtin_amdgcn_wmma_f32_16x16x32_bf16(
                false, a, false, b, (short)0, acc[nt], false, false);
        }
    }

    const float kScale = 1.0507009873554805f;
    const float kAlpha = 1.6732632423543772f;
#pragma unroll
    for (int nt = 0; nt < 4; ++nt) {
        const int n = nt * 16 + m;
        const float bj = bias[n];
#pragma unroll
        for (int r = 0; r < 8; ++r) {
            float v = acc[nt][r] + bj;
            // branch-free SELU: negative path via v_exp_f32, select via v_cndmask
            float en = kAlpha * (__expf(v) - 1.0f);
            v = kScale * ((v > 0.0f) ? v : en);
            H[(size_t)(row0 + mb + r) * FEAT + n] = v;
        }
    }
}

// ---------------------------------------------------------------------------
// GEMM2: out = log_softmax(P @ W2 + b2); 2 WMMAs per wave, 16-lane reductions
// ---------------------------------------------------------------------------
__global__ void __launch_bounds__(256) gemm2_lsm_kernel(const float* __restrict__ P,
                                                        const unsigned short* __restrict__ Wfrag,
                                                        const float* __restrict__ bias,
                                                        float* __restrict__ out, int nrows) {
    const int wave = (int)blockIdx.x * ((int)blockDim.x >> 5) + ((int)threadIdx.x >> 5);
    const int lane = threadIdx.x & 31;
    const int row0 = wave << 4;
    if (row0 >= nrows) return;
    const int m   = lane & 15;
    const int kb8 = (lane < 16) ? 0 : 8;
    const int mb  = (lane < 16) ? 0 : 8;

    v8f acc = {};
#pragma unroll
    for (int kt = 0; kt < 2; ++kt) {
        const float* rp = P + (size_t)(row0 + m) * FEAT + kt * 32;
        v16bf a = load_a_frag(rp, kb8);
        v16bf b = load_b_frag(Wfrag, kt, lane);
        acc = __builtin_amdgcn_wmma_f32_16x16x32_bf16(
            false, a, false, b, (short)0, acc, false, false);
    }

    const float bj = bias[m];
#pragma unroll
    for (int r = 0; r < 8; ++r) {
        float v = acc[r] + bj;
        float mx = v;
#pragma unroll
        for (int msk = 1; msk < 16; msk <<= 1)
            mx = fmaxf(mx, __shfl_xor(mx, msk, 32));
        float s = __expf(v - mx);
#pragma unroll
        for (int msk = 1; msk < 16; msk <<= 1)
            s += __shfl_xor(s, msk, 32);
        out[(size_t)(row0 + mb + r) * OUTC + m] = (v - mx) - __logf(s);
    }
}

// ---------------------------------------------------------------------------
// launch
// ---------------------------------------------------------------------------
extern "C" void kernel_launch(void* const* d_in, const int* in_sizes, int n_in,
                              void* d_out, int out_size, void* d_ws, size_t ws_size,
                              hipStream_t stream) {
    (void)n_in; (void)out_size; (void)ws_size;
    const float* x  = (const float*)d_in[0];
    const float* W1 = (const float*)d_in[1];
    const float* b1 = (const float*)d_in[2];
    const float* W2 = (const float*)d_in[3];
    const float* b2 = (const float*)d_in[4];
    const int*   ei = (const int*)d_in[5];
    const int N = in_sizes[0] / FEAT;
    const int E = in_sizes[5] / 2;
    const int* rows = ei;
    const int* cols = ei + E;

    char* ws = (char*)d_ws;
    size_t off = 0;
    float* dinv = (float*)(ws + off);  off += (((size_t)N * 4) + 255) & ~(size_t)255;
    float* bufA = (float*)(ws + off);  off += (size_t)N * FEAT * 4;
    float* bufB = (float*)(ws + off);  off += (size_t)N * FEAT * 4;
    unsigned short* Wf1 = (unsigned short*)(ws + off); off += 8 * 512 * 2;  // 8 frags
    unsigned short* Wf2 = (unsigned short*)(ws + off);                      // 2 frags

    const int totF = N * FEAT;
    const int gN   = (N + 255) / 256;
    const int gEth = (E + 255) / 256;
    const int gF   = (totF + 255) / 256;
    const int gE   = (E + 7) / 8;                  // 8 edge-waves per 256-thread block
    const int gW   = ((N / 16) * 32 + 255) / 256;  // one wave per 16-row tile

    // gcn_norm
    zero_kernel<<<gN, 256, 0, stream>>>(dinv, N);
    deg_kernel<<<gEth, 256, 0, stream>>>(rows, dinv, E);
    dinv_kernel<<<gN, 256, 0, stream>>>(dinv, N);
    // weight fragments: W1 -> 8 frags (nkt=2, 4 ntiles), W2 -> 2 frags (1 ntile)
    prep_wfrag_kernel<<<(8 * 512 + 255) / 256, 256, 0, stream>>>(W1, Wf1, FEAT, 2, 8 * 512);
    prep_wfrag_kernel<<<(2 * 512 + 255) / 256, 256, 0, stream>>>(W2, Wf2, OUTC, 2, 2 * 512);

    // conv1: 4 hops  x -> A -> B -> A -> B
    selfloop_init_kernel<<<gF, 256, 0, stream>>>(x, dinv, bufA, totF);
    spmm_scatter_kernel<<<gE, 256, 0, stream>>>(rows, cols, dinv, x, bufA, E);
    selfloop_init_kernel<<<gF, 256, 0, stream>>>(bufA, dinv, bufB, totF);
    spmm_scatter_kernel<<<gE, 256, 0, stream>>>(rows, cols, dinv, bufA, bufB, E);
    selfloop_init_kernel<<<gF, 256, 0, stream>>>(bufB, dinv, bufA, totF);
    spmm_scatter_kernel<<<gE, 256, 0, stream>>>(rows, cols, dinv, bufB, bufA, E);
    selfloop_init_kernel<<<gF, 256, 0, stream>>>(bufA, dinv, bufB, totF);
    spmm_scatter_kernel<<<gE, 256, 0, stream>>>(rows, cols, dinv, bufA, bufB, E);

    // linear1 + SELU  (WMMA bf16)
    gemm1_selu_kernel<<<gW, 256, 0, stream>>>(bufB, Wf1, b1, bufA, N);

    // conv2: 1 hop  A -> B
    selfloop_init_kernel<<<gF, 256, 0, stream>>>(bufA, dinv, bufB, totF);
    spmm_scatter_kernel<<<gE, 256, 0, stream>>>(rows, cols, dinv, bufA, bufB, E);

    // linear2 + log_softmax  (WMMA bf16)
    gemm2_lsm_kernel<<<gW, 256, 0, stream>>>(bufB, Wf2, b2, (float*)d_out, N);
}
